// MultiHeadGATLayerMerged_7318624273060
// MI455X (gfx1250) — matrix operand order repaired and checked by reading, not compiled, not run on hardware
//
#include <hip/hip_runtime.h>

#define BN   8
#define NN   1024
#define FIN  256
#define DOUT 256
#define ND   4
#define MROWS (BN*NN)   // 8192

typedef __attribute__((ext_vector_type(16))) __bf16 v16bf;
typedef __attribute__((ext_vector_type(8)))  float  v8f;

union FragAB { v16bf v; unsigned u[8]; uint4 q[2]; };
union PackD  { __bf16 h[8]; uint4 q; };

__device__ inline v8f wmma_bf16(v16bf a, v16bf b, v8f c) {
  return __builtin_amdgcn_wmma_f32_16x16x32_bf16(false, a, false, b, (short)0, c,
                                                 false, false);
}

// A fragment direct from global: matrix [m][k] row-major, row stride sk (elements).
// ISA 16-bit A layout: lanes 0-15 M=lane K={0..7,16..23}; lanes 16-31 K={8..15,24..31}.
__device__ inline v16bf fragA_g(const __bf16* __restrict__ g, int sk, int lane) {
  int r = lane & 15, h = lane >> 4;
  const __bf16* p = g + r*sk + h*8;
  FragAB f;
  f.q[0] = *(const uint4*)p;          // K = 8h .. 8h+7
  f.q[1] = *(const uint4*)(p + 16);   // K = 16+8h .. 16+8h+7
  return f.v;
}

// B fragment direct from global: matrix stored n-major [n][k], row stride sk.
// Layout: lanes 0-15 N=lane K={0..7,16..23}; lanes 16-31 K={8..15,24..31}.
__device__ inline v16bf fragB_g(const __bf16* __restrict__ g, int sk, int lane) {
  int n = lane & 15, h = lane >> 4;
  const __bf16* p = g + n*sk + h*8;
  FragAB f;
  f.q[0] = *(const uint4*)p;
  f.q[1] = *(const uint4*)(p + 16);
  return f.v;
}

// ---------------- small prep kernels ----------------

__global__ void k_cvt_x(const float* __restrict__ x, __bf16* __restrict__ xb, int n) {
  int i = blockIdx.x * blockDim.x + threadIdx.x;
  if (i < n) xb[i] = (__bf16)x[i];
}

// WsT[d][o][f] = bf16( W[d,0,f,o] + W[d,1,f,o] )   (n-major for direct B fragments)
__global__ void k_wsum(const float* __restrict__ W, __bf16* __restrict__ WsT) {
  int idx = blockIdx.x * blockDim.x + threadIdx.x;   // d*65536 + o*256 + f
  int d = idx >> 16;
  int o = (idx >> 8) & 255;
  int f = idx & 255;
  float w0 = W[(d*2 + 0)*65536 + f*256 + o];
  float w1 = W[(d*2 + 1)*65536 + f*256 + o];
  WsT[idx] = (__bf16)(w0 + w1);
}

// wsrc[d][f] = sum_o W[d,1,f,o]*a[d,1,o];  wdst with a[d,1,OUT+o]
__global__ void k_wvec(const float* __restrict__ W, const float* __restrict__ a,
                       float* __restrict__ wsrc, float* __restrict__ wdst) {
  int idx = blockIdx.x * blockDim.x + threadIdx.x;   // d*256 + f
  if (idx >= ND*FIN) return;
  int d = idx >> 8, f = idx & 255;
  const float* w1 = W + ((d*2 + 1)*256 + f)*256;
  const float* av = a + (d*2 + 1)*512;
  float s = 0.f, t = 0.f;
  for (int o = 0; o < 256; ++o) { float w = w1[o]; s += w*av[o]; t += w*av[256+o]; }
  wsrc[idx] = s; wdst[idx] = t;
}

// esrc[d][b][n] = x[b,n,:] . wsrc[d];  edst likewise
__global__ void k_evec(const float* __restrict__ x, const float* __restrict__ wsrc,
                       const float* __restrict__ wdst,
                       float* __restrict__ esrc, float* __restrict__ edst) {
  int idx = blockIdx.x * blockDim.x + threadIdx.x;   // d*8192 + b*1024 + n
  int d  = idx >> 13;
  int bn = idx & 8191;
  const float* xr = x + bn*256;
  const float* ws = wsrc + d*256;
  const float* wd = wdst + d*256;
  float s = 0.f, t = 0.f;
  for (int f = 0; f < 256; ++f) { float xv = xr[f]; s += xv*ws[f]; t += xv*wd[f]; }
  esrc[idx] = s; edst[idx] = t;
}

// ---------------- GEMM 1: WhT[d][o][m] = (xb @ WsT[d]^T) transposed-out ----------------
// M=8192 (m=b*1024+n), K=256 (f), N=256 (o). Wave tile 32x64, 8 accumulators.

__global__ void k_gemm_proj(const __bf16* __restrict__ xb, const __bf16* __restrict__ WsT,
                            __bf16* __restrict__ WhT) {
  int wave = threadIdx.x >> 5, lane = threadIdx.x & 31;
  int tile = blockIdx.x*8 + wave;         // 1024 tiles per direction
  int d  = blockIdx.y;
  int m0 = (tile >> 2) * 32;              // 256 row-tiles
  int n0 = (tile & 3) * 64;               // 4 col-tiles
  const __bf16* Ag = xb  + m0*FIN;                    // [m][f] stride 256
  const __bf16* Bg = WsT + d*DOUT*FIN + n0*FIN;       // [o][f] stride 256

  v8f acc[2][4] = {};
  for (int kk = 0; kk < FIN; kk += 32) {
    v16bf a0 = fragA_g(Ag + kk,          FIN, lane);
    v16bf a1 = fragA_g(Ag + 16*FIN + kk, FIN, lane);
#pragma unroll
    for (int c = 0; c < 4; ++c) {
      v16bf bv = fragB_g(Bg + c*16*FIN + kk, FIN, lane);
      acc[0][c] = wmma_bf16(a0, bv, acc[0][c]);
      acc[1][c] = wmma_bf16(a1, bv, acc[1][c]);
    }
  }

  // Transposed, packed store: WhT row o = n0+c*16+lane&15, cols m contiguous.
  int l = lane & 15, h = lane >> 4;
  __bf16* Dg = WhT + (size_t)d*DOUT*MROWS;
#pragma unroll
  for (int mi = 0; mi < 2; ++mi)
#pragma unroll
    for (int c = 0; c < 4; ++c) {
      PackD pk;
#pragma unroll
      for (int r = 0; r < 8; ++r) pk.h[r] = (__bf16)acc[mi][c][r];
      *(uint4*)(Dg + (n0 + c*16 + l)*MROWS + m0 + mi*16 + 8*h) = pk.q;
    }
}

// ---------------- fused mask-merge + leaky + softmax -> masked P planes ----------------

__global__ void k_softmax(const int* __restrict__ AU, const int* __restrict__ AD,
                          const int* __restrict__ AR, const int* __restrict__ AL,
                          const float* __restrict__ esrc, const float* __restrict__ edst,
                          __bf16* __restrict__ P) {
  int i = blockIdx.x, b = blockIdx.y, t = threadIdx.x;
  __shared__ float red[256];
  float es[4];
#pragma unroll
  for (int d = 0; d < 4; ++d) es[d] = esrc[d*8192 + b*1024 + i];

  float ev[4]; int msk[4];
  float mx = -3.4e38f;
#pragma unroll
  for (int q = 0; q < 4; ++q) {
    int j = t + 256*q;
    int au = AU[i*1024 + j] != 0;
    int ad = AD[i*1024 + j] != 0;
    int ar = AR[i*1024 + j] != 0;
    int al = AL[i*1024 + j] != 0;
    msk[q] = au | (ad << 1) | (ar << 2) | (al << 3);
    int dl = al ? 3 : (ar ? 2 : (ad ? 1 : (au ? 0 : -1)));
    float e = -3.4e38f;
    if (dl >= 0) {
      float v = es[dl] + edst[dl*8192 + b*1024 + j];
      e = v > 0.f ? v : 0.01f*v;       // leaky_relu(0.01)
    }
    ev[q] = e;
    mx = fmaxf(mx, e);
  }
  red[t] = mx; __syncthreads();
  for (int s = 128; s > 0; s >>= 1) { if (t < s) red[t] = fmaxf(red[t], red[t+s]); __syncthreads(); }
  mx = red[0]; __syncthreads();

  float al4[4]; float sum = 0.f;
#pragma unroll
  for (int q = 0; q < 4; ++q) {
    float p = msk[q] ? __expf(ev[q] - mx) : 0.f;
    al4[q] = p; sum += p;
  }
  red[t] = sum; __syncthreads();
  for (int s = 128; s > 0; s >>= 1) { if (t < s) red[t] += red[t+s]; __syncthreads(); }
  float inv = 1.f / red[0];

#pragma unroll
  for (int q = 0; q < 4; ++q) {
    int j = t + 256*q;
    float alpha = al4[q] * inv;
#pragma unroll
    for (int d = 0; d < 4; ++d)
      P[((d*8 + b)*NN + i)*NN + j] = ((msk[q] >> d) & 1) ? (__bf16)alpha : (__bf16)0.f;
  }
}

// ---------------- GEMM 2: out[b,i,o] = 0.5 * sum_d P[d,b] @ WhT[d]^T ----------------
// Per b: M=1024 (i), N=256 (o), K=1024*4 (j over 4 directions). Wave tile 32x64.

__global__ void k_gemm_out(const __bf16* __restrict__ P, const __bf16* __restrict__ WhT,
                           float* __restrict__ out) {
  int wave = threadIdx.x >> 5, lane = threadIdx.x & 31;
  int tile = blockIdx.x*8 + wave;         // 8 b * 32 itiles * 4 otiles = 1024
  int b  = tile >> 7;
  int i0 = ((tile >> 2) & 31) * 32;
  int o0 = (tile & 3) * 64;

  v8f acc[2][4] = {};
  for (int d = 0; d < 4; ++d) {
    const __bf16* Ag = P   + ((d*8 + b)*NN + i0)*NN;              // [i][j] stride 1024
    const __bf16* Bg = WhT + (size_t)d*DOUT*MROWS + o0*MROWS + b*NN; // [o][m] stride 8192
    for (int kk = 0; kk < NN; kk += 32) {
      __builtin_prefetch(Ag + kk + 64, 0, 1);   // global_prefetch_b8 two K tiles ahead
      v16bf a0 = fragA_g(Ag + kk,         NN, lane);
      v16bf a1 = fragA_g(Ag + 16*NN + kk, NN, lane);
#pragma unroll
      for (int c = 0; c < 4; ++c) {
        v16bf bv = fragB_g(Bg + c*16*MROWS + kk, MROWS, lane);
        acc[0][c] = wmma_bf16(a0, bv, acc[0][c]);
        acc[1][c] = wmma_bf16(a1, bv, acc[1][c]);
      }
    }
  }

  int l = lane & 15, h = lane >> 4;
#pragma unroll
  for (int mi = 0; mi < 2; ++mi)
#pragma unroll
    for (int c = 0; c < 4; ++c)
#pragma unroll
      for (int r = 0; r < 8; ++r)
        out[(b*NN + i0 + mi*16 + r + 8*h)*DOUT + o0 + c*16 + l] = 0.5f * acc[mi][c][r];
}

// ---------------- host launcher ----------------

extern "C" void kernel_launch(void* const* d_in, const int* in_sizes, int n_in,
                              void* d_out, int out_size, void* d_ws, size_t ws_size,
                              hipStream_t stream) {
  const float* x  = (const float*)d_in[0];
  const int*   AU = (const int*)d_in[1];
  const int*   AD = (const int*)d_in[2];
  const int*   AR = (const int*)d_in[3];
  const int*   AL = (const int*)d_in[4];
  const float* W  = (const float*)d_in[5];
  const float* a  = (const float*)d_in[6];
  float* out = (float*)d_out;
  char*  ws  = (char*)d_ws;

  size_t o_xb   = 0;                                  // 8192*256 bf16   = 4 MB
  size_t o_wst  = o_xb   + (size_t)MROWS*FIN*2;       // 4*256*256 bf16  = 512 KB
  size_t o_wsrc = o_wst  + (size_t)ND*FIN*DOUT*2;
  size_t o_wdst = o_wsrc + (size_t)ND*FIN*4;
  size_t o_esrc = o_wdst + (size_t)ND*FIN*4;
  size_t o_edst = o_esrc + (size_t)ND*BN*NN*4;
  size_t o_wht  = o_edst + (size_t)ND*BN*NN*4;        // 4*256*8192 bf16 = 16 MB
  size_t o_p    = o_wht  + (size_t)ND*DOUT*MROWS*2;   // 4*8*1024*1024 bf16 = 64 MB

  __bf16* xb   = (__bf16*)(ws + o_xb);
  __bf16* WsT  = (__bf16*)(ws + o_wst);
  float*  wsrc = (float*) (ws + o_wsrc);
  float*  wdst = (float*) (ws + o_wdst);
  float*  esrc = (float*) (ws + o_esrc);
  float*  edst = (float*) (ws + o_edst);
  __bf16* WhT  = (__bf16*)(ws + o_wht);
  __bf16* Pp   = (__bf16*)(ws + o_p);

  k_cvt_x<<<(MROWS*FIN + 255)/256, 256, 0, stream>>>(x, xb, MROWS*FIN);
  k_wsum <<<(ND*FIN*DOUT)/256,      256, 0, stream>>>(W, WsT);
  k_wvec <<<(ND*FIN + 255)/256,     256, 0, stream>>>(W, a, wsrc, wdst);
  k_gemm_proj<<<dim3(128, 4),       256, 0, stream>>>(xb, WsT, WhT);
  k_evec <<<(ND*BN*NN)/256,         256, 0, stream>>>(x, wsrc, wdst, esrc, edst);
  k_softmax<<<dim3(NN, BN),         256, 0, stream>>>(AU, AD, AR, AL, esrc, edst, Pp);
  k_gemm_out<<<128,                 256, 0, stream>>>(Pp, WhT, out);
}
